// SimplicialAttentionTransformer_36756330119901
// MI455X (gfx1250) — matrix operand order repaired and checked by reading, not compiled, not run on hardware
//
#include <hip/hip_runtime.h>
#include <stdint.h>

#define USE_TDM 1

typedef __bf16 bf16_t;
typedef __attribute__((ext_vector_type(16))) __bf16 v16bf;
typedef __attribute__((ext_vector_type(8)))  __bf16 v8bf;
typedef __attribute__((ext_vector_type(8)))  float  v8f;
typedef __attribute__((ext_vector_type(4)))  unsigned int uint32x4_t;
typedef __attribute__((ext_vector_type(8)))  int int32x8_t;
typedef __attribute__((ext_vector_type(4)))  int int32x4_t;

#define DD   512
#define TM   128
#define TN   128
#define KB   32
#define ASTR 40   // LDS row stride (bf16) for A tile: 80B rows, 16B-aligned chunks
#define BSTR 48   // LDS row stride for B tile: 96B rows, 32B-aligned v16bf loads

// ---------------- helpers ----------------

__device__ __forceinline__ unsigned short f2bf(float f) {
  unsigned u = __builtin_bit_cast(unsigned, f);
  u += 0x7FFFu + ((u >> 16) & 1u);        // round-to-nearest-even
  return (unsigned short)(u >> 16);
}

__device__ __forceinline__ unsigned fkey(float f) {   // order-preserving key
  unsigned u = __builtin_bit_cast(unsigned, f);
  return (u & 0x80000000u) ? ~u : (u | 0x80000000u);
}
__device__ __forceinline__ float fdekey(unsigned k) {
  unsigned u = (k & 0x80000000u) ? (k ^ 0x80000000u) : ~k;
  return __builtin_bit_cast(float, u);
}

// ---------------- elementwise ----------------

__global__ __launch_bounds__(256) void k_add3(const float* __restrict__ a,
                                              const float* __restrict__ b,
                                              const float* __restrict__ c,
                                              float* __restrict__ o, int n) {
  int i = blockIdx.x * blockDim.x + threadIdx.x;
  if (i < n) o[i] = a[i] + b[i] + c[i];
}

__global__ __launch_bounds__(256) void k_tobf16(const float* __restrict__ x,
                                                unsigned short* __restrict__ o, int n) {
  int i = blockIdx.x * blockDim.x + threadIdx.x;
  if (i < n) o[i] = f2bf(x[i]);
}

// convert W[k][n] (fp32) -> Wt[n][k] (bf16), nmat matrices of 512x512
__global__ __launch_bounds__(256) void k_wt(const float* __restrict__ W,
                                            unsigned short* __restrict__ Wt, int nmat) {
  int total = nmat * DD * DD;
  for (int i = blockIdx.x * blockDim.x + threadIdx.x; i < total;
       i += gridDim.x * blockDim.x) {
    int m = i >> 18;
    int r = i & (DD * DD - 1);
    int n = r >> 9;
    int k = r & (DD - 1);
    Wt[(size_t)m * DD * DD + (size_t)n * DD + k] =
        f2bf(W[(size_t)m * DD * DD + (size_t)k * DD + n]);
  }
}

// ---------------- WMMA GEMM:  C[M,512] = A[M,512](bf16) * Wt^T + bias ----------------
// Wt is [512 (n), 512 (k)] bf16 (pre-transposed weights).
// Double-buffered LDS; B tile staged by the Tensor Data Mover (wave 0 issues the
// descriptor, all waves s_wait_tensorcnt + barrier), A tile staged explicitly.

__global__ __launch_bounds__(256) void k_gemm_bias(
    const bf16_t* __restrict__ A, const bf16_t* __restrict__ Bt,
    const float* __restrict__ bias, float* __restrict__ C, int M) {
  __shared__ __align__(32) bf16_t As[2][TM * ASTR];
  __shared__ __align__(32) bf16_t Bs[2][TN * BSTR];

  const int tid  = threadIdx.x;
  const int lane = tid & 31;
  const int wave = tid >> 5;
  const int m0   = blockIdx.x * TM;
  const int n0   = blockIdx.y * TN;
  const int wm   = (wave & 3) * 32;   // 4 waves along M
  const int wn   = (wave >> 2) * 64;  // 2 waves along N
  const int mrow = lane & 15;
  const int half = lane >> 4;

  v8f acc[2][4];
#pragma unroll
  for (int i = 0; i < 2; ++i)
#pragma unroll
    for (int j = 0; j < 4; ++j)
#pragma unroll
      for (int r = 0; r < 8; ++r) acc[i][j][r] = 0.0f;

  auto stageA = [&](int buf, int k0) {
#pragma unroll
    for (int it = 0; it < 2; ++it) {
      int c   = tid + it * 256;
      int row = c >> 2;
      int kc  = (c & 3) * 8;
      uint4 av = make_uint4(0u, 0u, 0u, 0u);
      if (m0 + row < M)
        av = *(const uint4*)(A + (size_t)(m0 + row) * DD + k0 + kc);
      *(uint4*)(&As[buf][row * ASTR + kc]) = av;
    }
  };

  auto stageB = [&](int buf, int k0) {
#if USE_TDM
    if (wave == 0) {
      // D# group0: count=1 | lds_addr | global_addr[56:0] | type=2
      unsigned lds = (unsigned)(size_t)(void*)&Bs[buf][0];
      unsigned long long ga =
          (unsigned long long)(size_t)(const void*)(Bt + (size_t)n0 * DD + k0);
      uint32x4_t g0 = {1u, lds, (unsigned)ga,
                       ((unsigned)(ga >> 32) & 0x01FFFFFFu) | 0x80000000u};
      // D# group1: data_size=2B, pad_enable, pad_interval=16 DW (row=32 bf16),
      // pad_amount=8 DW (-> BSTR=48), tensor 512x512, tile 32x128, stride0=512
      int32x8_t g1 = {(int)((1u << 16) | (1u << 20) | (3u << 22) | (7u << 25)),
                      (int)(512u << 16),   // tensor_dim0[15:0] @ [31:16]
                      (int)(512u << 16),   // tensor_dim0 hi=0, tensor_dim1[15:0] @ [31:16]
                      (int)(32u << 16),    // tensor_dim1 hi=0, tile_dim0=32 @ [31:16]
                      128,                 // tile_dim1=128, tile_dim2=0
                      512,                 // tensor_dim0_stride[31:0]
                      0, 0};
      int32x4_t zz4 = {0, 0, 0, 0};
      int32x8_t zz8 = {0, 0, 0, 0, 0, 0, 0, 0};
      __builtin_amdgcn_tensor_load_to_lds(g0, g1, zz4, zz4, zz8, 0);
    }
#else
#pragma unroll
    for (int it = 0; it < 2; ++it) {
      int c   = tid + it * 256;
      int row = c >> 2;
      int kc  = (c & 3) * 8;
      uint4 bv = *(const uint4*)(Bt + (size_t)(n0 + row) * DD + k0 + kc);
      *(uint4*)(&Bs[buf][row * BSTR + kc]) = bv;
    }
#endif
  };

  auto compute = [&](int buf) {
    v16bf afr[2];
    v16bf bfr[4];
#pragma unroll
    for (int i = 0; i < 2; ++i) {
      const bf16_t* ap = &As[buf][(wm + i * 16 + mrow) * ASTR];
      v8bf lo = *(const v8bf*)(ap + half * 16);       // K = half*16 .. +8
      v8bf hi = *(const v8bf*)(ap + half * 16 + 8);   // K = half*16+8 .. +8
      afr[i] = __builtin_shufflevector(lo, hi, 0, 1, 2, 3, 4, 5, 6, 7,
                                       8, 9, 10, 11, 12, 13, 14, 15);
    }
#pragma unroll
    for (int j = 0; j < 4; ++j) {
      const bf16_t* bp = &Bs[buf][(wn + j * 16 + mrow) * BSTR];
      bfr[j] = *(const v16bf*)(bp + half * 16);        // K = half*16 .. +16
    }
#pragma unroll
    for (int i = 0; i < 2; ++i)
#pragma unroll
      for (int j = 0; j < 4; ++j)
        acc[i][j] = __builtin_amdgcn_wmma_f32_16x16x32_bf16(
            false, afr[i], false, bfr[j], (short)0, acc[i][j], false, false);
  };

  // prologue: stage tile 0
  stageA(0, 0);
  stageB(0, 0);
  __builtin_amdgcn_s_wait_tensorcnt(0);
  __syncthreads();

  for (int k0 = 0; k0 < DD; k0 += KB) {
    int buf = (k0 / KB) & 1;
    if (k0 + KB < DD) {
      stageA(buf ^ 1, k0 + KB);
      stageB(buf ^ 1, k0 + KB);
    }
    if (k0 + 2 * KB < DD && tid < TM && m0 + tid < M)
      __builtin_prefetch(A + (size_t)(m0 + tid) * DD + k0 + 2 * KB, 0, 1);
    compute(buf);
    __builtin_amdgcn_s_wait_tensorcnt(0);
    __syncthreads();
  }

#pragma unroll
  for (int i = 0; i < 2; ++i)
#pragma unroll
    for (int j = 0; j < 4; ++j) {
      int col = n0 + wn + j * 16 + mrow;
      float bv = bias[col];
#pragma unroll
      for (int r = 0; r < 8; ++r) {
        int row = m0 + wm + i * 16 + half * 8 + r;
        if (row < M) C[(size_t)row * DD + col] = acc[i][j][r] + bv;
      }
    }
}

// ---------------- attention edge kernels ----------------

__global__ __launch_bounds__(256) void k_scores(const float* __restrict__ q,
                                                const float* __restrict__ kk,
                                                const int* __restrict__ ei, int E,
                                                float* __restrict__ scores) {
  int idx = blockIdx.x * blockDim.x + threadIdx.x;
  if (idx >= E * 8) return;
  int e = idx >> 3, h = idx & 7;
  int src = ei[e], dst = ei[E + e];
  const float4* qp = (const float4*)(q + (size_t)dst * DD + h * 64);
  const float4* kp = (const float4*)(kk + (size_t)src * DD + h * 64);
  float s = 0.f;
#pragma unroll
  for (int t = 0; t < 16; ++t) {
    float4 a = qp[t], b = kp[t];
    s += a.x * b.x + a.y * b.y + a.z * b.z + a.w * b.w;
  }
  scores[idx] = s * 0.125f;
}

__global__ __launch_bounds__(256) void k_redmax(const float* __restrict__ s, int E,
                                                unsigned* __restrict__ rmax) {
  int T = gridDim.x * blockDim.x;
  int tid = blockIdx.x * blockDim.x + threadIdx.x;
  int h = tid & 7;
  float m = -3.4e38f;
  for (int e = tid >> 3; e < E; e += (T >> 3)) m = fmaxf(m, s[e * 8 + h]);
  atomicMax(&rmax[h], fkey(m));
}

__global__ __launch_bounds__(256) void k_sumexp(const float* __restrict__ s, int E,
                                                const unsigned* __restrict__ rmax,
                                                float* __restrict__ sum) {
  int T = gridDim.x * blockDim.x;
  int tid = blockIdx.x * blockDim.x + threadIdx.x;
  int h = tid & 7;
  float mx = fdekey(rmax[h]);
  float acc = 0.f;
  for (int e = tid >> 3; e < E; e += (T >> 3)) acc += __expf(s[e * 8 + h] - mx);
  atomicAdd(&sum[h], acc);
}

__global__ __launch_bounds__(256) void k_scatter(
    const float* __restrict__ v, const float* __restrict__ s,
    const unsigned* __restrict__ rmax, const float* __restrict__ sum,
    const int* __restrict__ ei, int E, float* __restrict__ ctx) {
  int total = E * DD;
  int stride = gridDim.x * blockDim.x;
  for (int idx = blockIdx.x * blockDim.x + threadIdx.x; idx < total; idx += stride) {
    int e = idx >> 9, d = idx & (DD - 1), h = d >> 6;
    int src = ei[e], dst = ei[E + e];
    float w = __expf(s[e * 8 + h] - fdekey(rmax[h])) / sum[h];
    atomicAdd(&ctx[(size_t)dst * DD + d], w * v[(size_t)src * DD + d]);
  }
}

// x = LayerNorm(o + x) * g + bt   (one 128-thread block per row)
__global__ __launch_bounds__(128) void k_ln(const float* __restrict__ o,
                                            float* __restrict__ x,
                                            const float* __restrict__ g,
                                            const float* __restrict__ bt) {
  __shared__ float s1[128], s2[128];
  int row = blockIdx.x;
  int tid = threadIdx.x;
  const float* op = o + (size_t)row * DD;
  float* xp = x + (size_t)row * DD;
  float vals[4];
  float a = 0.f, b = 0.f;
#pragma unroll
  for (int j = 0; j < 4; ++j) {
    int c = tid + j * 128;
    float t = op[c] + xp[c];
    vals[j] = t;
    a += t;
    b += t * t;
  }
  s1[tid] = a; s2[tid] = b;
  __syncthreads();
  for (int st = 64; st > 0; st >>= 1) {
    if (tid < st) { s1[tid] += s1[tid + st]; s2[tid] += s2[tid + st]; }
    __syncthreads();
  }
  float mean = s1[0] * (1.f / 512.f);
  float var  = s2[0] * (1.f / 512.f) - mean * mean;
  float rstd = rsqrtf(var + 1e-5f);
#pragma unroll
  for (int j = 0; j < 4; ++j) {
    int c = tid + j * 128;
    xp[c] = (vals[j] - mean) * rstd * g[c] + bt[c];
  }
}

// ---------------- host orchestration ----------------

extern "C" void kernel_launch(void* const* d_in, const int* in_sizes, int n_in,
                              void* d_out, int out_size, void* d_ws, size_t ws_size,
                              hipStream_t stream) {
  (void)in_sizes; (void)n_in; (void)out_size;
  const int NV_ = 20000, NE_ = 40000, NT_ = 30000;
  const int Ns[3] = {NV_, NE_, NT_};
  const int Es[3] = {160000, 160000, 120000};
  const size_t MATS = (size_t)DD * DD;

  char* wsb = (char*)d_ws;
  size_t off = 0;
  auto carve = [&](size_t bytes) -> char* {
    char* p = wsb + off;
    off += (bytes + 255) & ~(size_t)255;
    return p;
  };
  unsigned short* wtAll = (unsigned short*)carve(27 * MATS * 2);        // bf16 W^T
  float*          xf    = (float*)carve((size_t)NE_ * DD * 4);          // fp32 activations
  unsigned short* xb    = (unsigned short*)carve((size_t)NE_ * DD * 2); // bf16 A operand
  float*          qb    = (float*)carve((size_t)NE_ * DD * 4);
  float*          kb    = (float*)carve((size_t)NE_ * DD * 4);
  float*          vb    = (float*)carve((size_t)NE_ * DD * 4);
  float*          cx    = (float*)carve((size_t)NE_ * DD * 4);
  float*          sc    = (float*)carve((size_t)160000 * 8 * 4);
  unsigned*       red   = (unsigned*)carve(256); // [0..7] max keys, [8..15] sumexp
  if (off > ws_size) return;

  const float* headB = (const float*)d_in[22];

  // weight transpose + bf16 convert: v(8 mats), e(8), t(8), head(3)
  k_wt<<<8192, 256, 0, stream>>>((const float*)d_in[9],  wtAll +  0 * MATS, 8);
  k_wt<<<8192, 256, 0, stream>>>((const float*)d_in[13], wtAll +  8 * MATS, 8);
  k_wt<<<8192, 256, 0, stream>>>((const float*)d_in[17], wtAll + 16 * MATS, 8);
  k_wt<<<3072, 256, 0, stream>>>((const float*)d_in[21], wtAll + 24 * MATS, 3);

  for (int ent = 0; ent < 3; ++ent) {
    int N = Ns[ent], E = Es[ent];
    int nElem = N * DD;
    const float* feat = (const float*)d_in[0 + ent];
    const float* emb  = (const float*)d_in[3 + ent];
    const float* pos  = (const float*)d_in[6 + ent];
    const float* Bb   = (const float*)d_in[10 + 4 * ent];
    const float* Gb   = (const float*)d_in[11 + 4 * ent];
    const float* Btb  = (const float*)d_in[12 + 4 * ent];
    const int*   ei   = (const int*)d_in[23 + ent];

    k_add3<<<(nElem + 255) / 256, 256, 0, stream>>>(emb, pos, feat, xf, nElem);

    dim3 gg((N + TM - 1) / TM, DD / TN);

    for (int l = 0; l < 2; ++l) {
      const unsigned short* wt = wtAll + ((size_t)ent * 8 + (size_t)l * 4) * MATS;
      const float* bias = Bb + (size_t)(l * 4) * DD;

      k_tobf16<<<(nElem + 255) / 256, 256, 0, stream>>>(xf, xb, nElem);
      k_gemm_bias<<<gg, 256, 0, stream>>>((const bf16_t*)xb, (const bf16_t*)(wt + 0 * MATS), bias + 0 * DD, qb, N);
      k_gemm_bias<<<gg, 256, 0, stream>>>((const bf16_t*)xb, (const bf16_t*)(wt + 1 * MATS), bias + 1 * DD, kb, N);
      k_gemm_bias<<<gg, 256, 0, stream>>>((const bf16_t*)xb, (const bf16_t*)(wt + 2 * MATS), bias + 2 * DD, vb, N);

      k_scores<<<(E * 8 + 255) / 256, 256, 0, stream>>>(qb, kb, ei, E, sc);
      (void)hipMemsetAsync(red, 0, 64, stream);
      k_redmax<<<64, 256, 0, stream>>>(sc, E, red);
      k_sumexp<<<64, 256, 0, stream>>>(sc, E, red, (float*)(red + 8));
      (void)hipMemsetAsync(cx, 0, (size_t)nElem * 4, stream);
      k_scatter<<<4096, 256, 0, stream>>>(vb, sc, red, (const float*)(red + 8), ei, E, cx);

      k_tobf16<<<(nElem + 255) / 256, 256, 0, stream>>>(cx, xb, nElem);
      k_gemm_bias<<<gg, 256, 0, stream>>>((const bf16_t*)xb, (const bf16_t*)(wt + 3 * MATS), bias + 3 * DD, qb, N);
      k_ln<<<N, 128, 0, stream>>>(qb, xf, Gb + (size_t)l * DD, Btb + (size_t)l * DD);
    }

    size_t outOff = 0;
    if (ent == 1) outOff = (size_t)NV_ * DD;
    if (ent == 2) outOff = (size_t)(NV_ + NE_) * DD;
    k_tobf16<<<(nElem + 255) / 256, 256, 0, stream>>>(xf, xb, nElem);
    k_gemm_bias<<<gg, 256, 0, stream>>>((const bf16_t*)xb,
                                        (const bf16_t*)(wtAll + (24 + (size_t)ent) * MATS),
                                        headB + (size_t)ent * DD,
                                        (float*)d_out + outOff, N);
  }
}
